// OmniMoE_83150566850562
// MI455X (gfx1250) — matrix operand order repaired and checked by reading, not compiled, not run on hardware
//
#include <hip/hip_runtime.h>
#include <hip/hip_bf16.h>

// ---------------------------------------------------------------------------
// Types for CDNA5 WMMA
// ---------------------------------------------------------------------------
typedef __bf16 bf16_t;
typedef __bf16 v8bf  __attribute__((ext_vector_type(8)));
typedef __bf16 v16bf __attribute__((ext_vector_type(16)));
typedef float  v8f   __attribute__((ext_vector_type(8)));

union FragU { v16bf v; v8bf h[2]; };

// fp32 -> bf16 round-to-nearest-even (bit trick)
static __device__ __forceinline__ bf16_t f2bf(float f) {
    unsigned int u = __float_as_uint(f);
    unsigned int r = u + 0x7FFFu + ((u >> 16) & 1u);
    unsigned short hs = (unsigned short)(r >> 16);
    bf16_t b;
    __builtin_memcpy(&b, &hs, sizeof(hs));
    return b;
}

// ---------------------------------------------------------------------------
// CDNA5 async global->LDS copy (16B per lane), tracked by ASYNCcnt.
// Generic pointers to LDS carry the LDS byte offset in their low 32 bits
// (ISA 10.2).
// ---------------------------------------------------------------------------
static __device__ __forceinline__ void async_copy_b128(const bf16_t* gptr,
                                                       const bf16_t* lptr) {
    unsigned int laddr = (unsigned int)(unsigned long long)lptr;
    asm volatile("global_load_async_to_lds_b128 %0, %1, off"
                 :: "v"(laddr), "v"(gptr) : "memory");
}

static __device__ __forceinline__ void wait_async0() {
    asm volatile("s_wait_asynccnt 0x0" ::: "memory");
}

// ---------------------------------------------------------------------------
// Problem constants (from reference setup_inputs)
// ---------------------------------------------------------------------------
static constexpr int T_TOK = 8192;   // 4 * 2048
static constexpr int H_DIM = 1024;
static constexpr int I_DIM = 4096;
static constexpr int S_DIM = 128;
static constexpr int K_TOP = 16;

// GEMM tiling
static constexpr int BM = 128;
static constexpr int BN = 64;
static constexpr int BK = 64;
static constexpr int AST = BK + 8;   // padded LDS stride (bf16 elems) -> 144B rows
static constexpr int WST = BK + 8;

// ---------------------------------------------------------------------------
// fp32 -> bf16 conversion kernel
// ---------------------------------------------------------------------------
__global__ __launch_bounds__(256) void cvt_f32_bf16(const float* __restrict__ src,
                                                    bf16_t* __restrict__ dst, int n) {
    int i = blockIdx.x * 256 + threadIdx.x;
    if (i < n) dst[i] = f2bf(src[i]);
}

// ---------------------------------------------------------------------------
// LDS fragment loaders matching CDNA5 WMMA 16-bit layouts
// ---------------------------------------------------------------------------
// A (16x32, MxK): lanes 0-15 row m, K={kk+0..7, kk+16..23}; lanes 16-31 K=+8
static __device__ __forceinline__ v16bf load_fragA(const bf16_t* As, int rowBase,
                                                   int kk, int lane) {
    int hi = lane >> 4;
    int r  = rowBase + (lane & 15);
    const bf16_t* p = As + r * AST + kk + hi * 8;
    FragU f;
    f.h[0] = *(const v8bf*)(p);
    f.h[1] = *(const v8bf*)(p + 16);
    return f.v;
}

// B (32x16, KxN): lanes 0-15 col n, K={kk+0..7, kk+16..23}; lanes 16-31 K=+8
// (column n of B == row n of W[N,K] -> contiguous)
static __device__ __forceinline__ v16bf load_fragB(const bf16_t* Ws, int colBase,
                                                   int kk, int lane) {
    int hi = lane >> 4;
    int n  = colBase + (lane & 15);
    const bf16_t* p = Ws + n * WST + kk + hi * 8;
    FragU f;
    f.h[0] = *(const v8bf*)(p);
    f.h[1] = *(const v8bf*)(p + 16);
    return f.v;
}

static __device__ __forceinline__ v8f wmma_bf16(v16bf a, v16bf b, v8f c) {
    return __builtin_amdgcn_wmma_f32_16x16x32_bf16(
        /*neg_a=*/false, a, /*neg_b=*/false, b,
        /*c_mod=*/(short)0, c, /*reuse_a=*/false, /*reuse_b=*/false);
}

// ---------------------------------------------------------------------------
// Generic bf16 GEMM:  C[t,n] = sum_k A[t,k] * W[n,k]
// MODE 0: single GEMM, fp32 store to Cf
// MODE 1: dual GEMM (W0=gate, W1=up), epilogue silu(g)*u -> bf16 store to Cb
// Grid: (N/BN, M/BM), block 256 threads = 8 waves (4 M-waves x 2 N-waves),
// each wave computes a 32x32 output tile per GEMM.
// Double-buffered LDS, async global->LDS staging (ASYNCcnt).
// ---------------------------------------------------------------------------
template <int MODE>
__global__ __launch_bounds__(256)
void gemm_bf16_wmma(const bf16_t* __restrict__ A,
                    const bf16_t* __restrict__ W0,
                    const bf16_t* __restrict__ W1,
                    float* __restrict__ Cf,
                    bf16_t* __restrict__ Cb,
                    int M, int N, int K) {
    constexpr int NW = (MODE == 1) ? 2 : 1;
    __shared__ bf16_t As[2][BM * AST];
    __shared__ bf16_t Ws[2][NW][BN * WST];

    const int tid  = threadIdx.x;
    const int lane = tid & 31;
    const int wv   = tid >> 5;
    const int wM   = wv & 3;        // 0..3
    const int wN   = wv >> 2;       // 0..1
    const int bM   = blockIdx.y * BM;
    const int bN   = blockIdx.x * BN;

    // issue async copies for one K-tile into LDS stage `st`
    auto stage_tile = [&](int st, int k0) {
        // A tile: 128 x 64 bf16 = 1024 x 16B chunks / 256 threads
        #pragma unroll
        for (int r = 0; r < 4; ++r) {
            int c   = tid + r * 256;
            int row = c >> 3;
            int col = c & 7;
            async_copy_b128(A + (size_t)(bM + row) * K + k0 + col * 8,
                            &As[st][row * AST + col * 8]);
        }
        // W tile(s): 64 x 64 bf16 = 512 x 16B chunks / 256 threads
        #pragma unroll
        for (int r = 0; r < 2; ++r) {
            int c   = tid + r * 256;
            int row = c >> 3;
            int col = c & 7;
            async_copy_b128(W0 + (size_t)(bN + row) * K + k0 + col * 8,
                            &Ws[st][0][row * WST + col * 8]);
            if (MODE == 1) {
                async_copy_b128(W1 + (size_t)(bN + row) * K + k0 + col * 8,
                                &Ws[st][NW - 1][row * WST + col * 8]);
            }
        }
    };

    v8f acc0[2][2] = {};
    v8f acc1[2][2] = {};

    // prologue: stage tile 0
    stage_tile(0, 0);
    wait_async0();
    __syncthreads();

    int buf = 0;
    for (int k0 = 0; k0 < K; k0 += BK) {
        // overlap: stream next tile into the other buffer while computing
        if (k0 + BK < K) stage_tile(buf ^ 1, k0 + BK);

        const bf16_t* Ab  = As[buf];
        const bf16_t* Wb0 = Ws[buf][0];
        const bf16_t* Wb1 = Ws[buf][NW - 1];

        #pragma unroll
        for (int kk = 0; kk < BK; kk += 32) {
            v16bf a0 = load_fragA(Ab, wM * 32,      kk, lane);
            v16bf a1 = load_fragA(Ab, wM * 32 + 16, kk, lane);
            v16bf b0 = load_fragB(Wb0, wN * 32,      kk, lane);
            v16bf b1 = load_fragB(Wb0, wN * 32 + 16, kk, lane);
            acc0[0][0] = wmma_bf16(a0, b0, acc0[0][0]);
            acc0[0][1] = wmma_bf16(a0, b1, acc0[0][1]);
            acc0[1][0] = wmma_bf16(a1, b0, acc0[1][0]);
            acc0[1][1] = wmma_bf16(a1, b1, acc0[1][1]);
            if (MODE == 1) {
                v16bf c0 = load_fragB(Wb1, wN * 32,      kk, lane);
                v16bf c1 = load_fragB(Wb1, wN * 32 + 16, kk, lane);
                acc1[0][0] = wmma_bf16(a0, c0, acc1[0][0]);
                acc1[0][1] = wmma_bf16(a0, c1, acc1[0][1]);
                acc1[1][0] = wmma_bf16(a1, c0, acc1[1][0]);
                acc1[1][1] = wmma_bf16(a1, c1, acc1[1][1]);
            }
        }
        // all lanes' LDS reads for this tile completed before their consuming
        // WMMAs (compiler-inserted s_wait_dscnt); now ensure next tile landed.
        wait_async0();
        __syncthreads();
        buf ^= 1;
    }

    // ---- epilogue (C 16x16 f32 layout: VGPR v -> row v + 8*(lane>>4)) -----
    const int hi = lane >> 4;
    const int ln = lane & 15;
    #pragma unroll
    for (int i = 0; i < 2; ++i) {
        #pragma unroll
        for (int j = 0; j < 2; ++j) {
            int rowBase = bM + wM * 32 + i * 16 + hi * 8;
            int cg      = bN + wN * 32 + j * 16 + ln;
            #pragma unroll
            for (int v = 0; v < 8; ++v) {
                int rg = rowBase + v;
                if (MODE == 0) {
                    Cf[(size_t)rg * N + cg] = acc0[i][j][v];
                } else {
                    float g = acc0[i][j][v];
                    float u = acc1[i][j][v];
                    float s = g / (1.0f + __expf(-g));   // silu
                    Cb[(size_t)rg * N + cg] = f2bf(s * u);
                }
            }
        }
    }
}

// ---------------------------------------------------------------------------
// BatchNorm statistics: per-feature mean/rstd over T for gx (blocks 0-127)
// and gy (blocks 128-255). stats layout: [mx(128) rx(128) my(128) ry(128)]
// ---------------------------------------------------------------------------
__global__ __launch_bounds__(256) void bn_stats_kernel(const float* __restrict__ gx,
                                                       const float* __restrict__ gy,
                                                       float* __restrict__ stats) {
    int b   = blockIdx.x;
    int mat = b >> 7;
    int col = b & 127;
    const float* src = mat ? gy : gx;
    float s = 0.f, sq = 0.f;
    for (int t = threadIdx.x; t < T_TOK; t += 256) {
        float v = src[(size_t)t * S_DIM + col];
        s += v;
        sq += v * v;
    }
    __shared__ float sh[256], shq[256];
    sh[threadIdx.x]  = s;
    shq[threadIdx.x] = sq;
    __syncthreads();
    for (int o = 128; o > 0; o >>= 1) {
        if (threadIdx.x < o) {
            sh[threadIdx.x]  += sh[threadIdx.x + o];
            shq[threadIdx.x] += shq[threadIdx.x + o];
        }
        __syncthreads();
    }
    if (threadIdx.x == 0) {
        float mean = sh[0] * (1.0f / T_TOK);
        float var  = shq[0] * (1.0f / T_TOK) - mean * mean;
        stats[mat * 256 + col]       = mean;
        stats[mat * 256 + 128 + col] = rsqrtf(var + 1e-5f);
    }
}

// ---------------------------------------------------------------------------
// Per-token: batchnorm -> log-softmax -> top-16 per axis -> exact top-16 of
// the S*S sum via the 16x16 candidate set (any top-k pair must have both
// coordinates in their per-axis top-k).
// ---------------------------------------------------------------------------
static __device__ void top16_logsoftmax(const float* __restrict__ g,
                                        const float* __restrict__ mean,
                                        const float* __restrict__ rstd,
                                        float* tv, int* ti) {
    float z[S_DIM];
    float m = -3.4e38f;
    #pragma unroll 4
    for (int s = 0; s < S_DIM; ++s) {
        float v = (g[s] - mean[s]) * rstd[s];
        z[s] = v;
        m = fmaxf(m, v);
    }
    float se = 0.f;
    #pragma unroll 4
    for (int s = 0; s < S_DIM; ++s) se += __expf(z[s] - m);
    float lse = m + __logf(se);
    #pragma unroll
    for (int k = 0; k < K_TOP; ++k) { tv[k] = -3.4e38f; ti[k] = 0; }
    for (int s = 0; s < S_DIM; ++s) {
        float v = z[s] - lse;
        if (v > tv[K_TOP - 1]) {
            int p = K_TOP - 1;
            while (p > 0 && v > tv[p - 1]) {
                tv[p] = tv[p - 1];
                ti[p] = ti[p - 1];
                --p;
            }
            tv[p] = v;
            ti[p] = s;
        }
    }
}

__global__ __launch_bounds__(256) void router_topk_kernel(const float* __restrict__ gx,
                                                          const float* __restrict__ gy,
                                                          const float* __restrict__ stats,
                                                          int* __restrict__ oidx,
                                                          float* __restrict__ ow) {
    int t = blockIdx.x * 256 + threadIdx.x;
    if (t >= T_TOK) return;

    float txv[K_TOP], tyv[K_TOP];
    int   txi[K_TOP], tyi[K_TOP];
    top16_logsoftmax(gx + (size_t)t * S_DIM, stats + 0,   stats + 128, txv, txi);
    top16_logsoftmax(gy + (size_t)t * S_DIM, stats + 256, stats + 384, tyv, tyi);

    float cv[K_TOP];
    int   ci[K_TOP];
    #pragma unroll
    for (int k = 0; k < K_TOP; ++k) { cv[k] = -3.4e38f; ci[k] = 0; }
    for (int i = 0; i < K_TOP; ++i) {
        for (int j = 0; j < K_TOP; ++j) {
            float v = txv[i] + tyv[j];
            if (v > cv[K_TOP - 1]) {
                int idx = txi[i] * S_DIM + tyi[j];
                int p = K_TOP - 1;
                while (p > 0 && v > cv[p - 1]) {
                    cv[p] = cv[p - 1];
                    ci[p] = ci[p - 1];
                    --p;
                }
                cv[p] = v;
                ci[p] = idx;
            }
        }
    }
    #pragma unroll
    for (int k = 0; k < K_TOP; ++k) {
        oidx[(size_t)t * K_TOP + k] = ci[k];
        ow[(size_t)t * K_TOP + k]   = __expf(cv[k]);
    }
}

// ---------------------------------------------------------------------------
// Expert branch: one wave32 per token.
//   score_e = dot(up_embed[e], x);  out[t] += sum_e w_e*score_e * down_embed[e]
// Tables are L2-resident (2 x 64 MB << 192 MB L2).
// ---------------------------------------------------------------------------
__global__ __launch_bounds__(256) void expert_kernel(const float* __restrict__ x,
                                                     const float* __restrict__ ue,
                                                     const float* __restrict__ de,
                                                     const int* __restrict__ idx,
                                                     const float* __restrict__ w,
                                                     float* __restrict__ out) {
    int wave = blockIdx.x * 8 + (threadIdx.x >> 5);
    int lane = threadIdx.x & 31;
    int t = wave;
    if (t >= T_TOK) return;

    float xr[32];
    #pragma unroll
    for (int i = 0; i < 32; ++i) xr[i] = x[(size_t)t * H_DIM + i * 32 + lane];

    float acc[32];
    #pragma unroll
    for (int i = 0; i < 32; ++i) acc[i] = 0.f;

    for (int e = 0; e < K_TOP; ++e) {
        int   ex = idx[(size_t)t * K_TOP + e];
        float wt = w[(size_t)t * K_TOP + e];
        const float* up = ue + (size_t)ex * H_DIM;
        const float* dn = de + (size_t)ex * H_DIM;
        float p = 0.f;
        #pragma unroll
        for (int i = 0; i < 32; ++i) p += xr[i] * up[i * 32 + lane];
        #pragma unroll
        for (int off = 16; off > 0; off >>= 1) p += __shfl_xor(p, off, 32);
        float s = wt * p;
        #pragma unroll
        for (int i = 0; i < 32; ++i) acc[i] += s * dn[i * 32 + lane];
    }
    #pragma unroll
    for (int i = 0; i < 32; ++i) out[(size_t)t * H_DIM + i * 32 + lane] += acc[i];
}

// ---------------------------------------------------------------------------
// Host orchestration
// ---------------------------------------------------------------------------
extern "C" void kernel_launch(void* const* d_in, const int* in_sizes, int n_in,
                              void* d_out, int out_size, void* d_ws, size_t ws_size,
                              hipStream_t stream) {
    const float* x_f32 = (const float*)d_in[0];  // [T, H]
    const float* wg    = (const float*)d_in[1];  // [I, H]
    const float* wu    = (const float*)d_in[2];  // [I, H]
    const float* wd    = (const float*)d_in[3];  // [H, I]
    const float* rx    = (const float*)d_in[4];  // [S, H]
    const float* ry    = (const float*)d_in[5];  // [S, H]
    const float* ue    = (const float*)d_in[6];  // [E, H]
    const float* de    = (const float*)d_in[7];  // [E, H]

    char* wp = (char*)d_ws;
    auto alloc = [&](size_t bytes) -> char* {
        char* p = wp;
        wp += (bytes + 255) & ~(size_t)255;
        return p;
    };
    bf16_t* xb  = (bf16_t*)alloc((size_t)T_TOK * H_DIM * 2);
    bf16_t* wgb = (bf16_t*)alloc((size_t)I_DIM * H_DIM * 2);
    bf16_t* wub = (bf16_t*)alloc((size_t)I_DIM * H_DIM * 2);
    bf16_t* wdb = (bf16_t*)alloc((size_t)H_DIM * I_DIM * 2);
    bf16_t* rxb = (bf16_t*)alloc((size_t)S_DIM * H_DIM * 2);
    bf16_t* ryb = (bf16_t*)alloc((size_t)S_DIM * H_DIM * 2);
    bf16_t* hb  = (bf16_t*)alloc((size_t)T_TOK * I_DIM * 2);
    float*  gx  = (float*) alloc((size_t)T_TOK * S_DIM * 4);
    float*  gy  = (float*) alloc((size_t)T_TOK * S_DIM * 4);
    float*  st  = (float*) alloc(512 * 4);
    int*    tix = (int*)   alloc((size_t)T_TOK * K_TOP * 4);
    float*  tw  = (float*) alloc((size_t)T_TOK * K_TOP * 4);

    float* out = (float*)d_out;

    // 1) fp32 -> bf16 conversions
    auto cvt = [&](const float* s, bf16_t* d, int n) {
        cvt_f32_bf16<<<(n + 255) / 256, 256, 0, stream>>>(s, d, n);
    };
    cvt(x_f32, xb,  T_TOK * H_DIM);
    cvt(wg,    wgb, I_DIM * H_DIM);
    cvt(wu,    wub, I_DIM * H_DIM);
    cvt(wd,    wdb, H_DIM * I_DIM);
    cvt(rx,    rxb, S_DIM * H_DIM);
    cvt(ry,    ryb, S_DIM * H_DIM);

    // 2) fused gate/up GEMM + SiLU*up epilogue -> h (bf16) [T, I]
    gemm_bf16_wmma<1><<<dim3(I_DIM / BN, T_TOK / BM), 256, 0, stream>>>(
        xb, wgb, wub, nullptr, hb, T_TOK, I_DIM, H_DIM);

    // 3) router GEMMs -> gx, gy (fp32) [T, S]
    gemm_bf16_wmma<0><<<dim3(S_DIM / BN, T_TOK / BM), 256, 0, stream>>>(
        xb, rxb, nullptr, gx, nullptr, T_TOK, S_DIM, H_DIM);
    gemm_bf16_wmma<0><<<dim3(S_DIM / BN, T_TOK / BM), 256, 0, stream>>>(
        xb, ryb, nullptr, gy, nullptr, T_TOK, S_DIM, H_DIM);

    // 4) down GEMM -> d_out (fp32) [T, H]  (base for expert accumulation)
    gemm_bf16_wmma<0><<<dim3(H_DIM / BN, T_TOK / BM), 256, 0, stream>>>(
        hb, wdb, nullptr, out, nullptr, T_TOK, H_DIM, I_DIM);

    // 5) router postprocess: batchnorm stats, then top-k selection
    bn_stats_kernel<<<2 * S_DIM, 256, 0, stream>>>(gx, gy, st);
    router_topk_kernel<<<T_TOK / 256, 256, 0, stream>>>(gx, gy, st, tix, tw);

    // 6) expert gather/dot/combine, accumulate into d_out
    expert_kernel<<<T_TOK / 8, 256, 0, stream>>>(x_f32, ue, de, tix, tw, out);
}
